// MSNN_23510650979032
// MI455X (gfx1250) — matrix operand order repaired and checked by reading, not compiled, not run on hardware
//
#include <hip/hip_runtime.h>

// ---------------------------------------------------------------------------
// SNN (2-layer LIF) on MI455X / gfx1250.
//   cur1 = x @ w1^T + b1            (once,  256x4096x2048, bf16 3-product split)
//   50x:  LIF1 -> spk1 (exact {0,1} bf16)
//         cur2 = spk1 @ w2^T + b2   (256x1024x4096, bf16 WMMA, hi/lo split,
//                                    B staged via global_load_async_to_lds_b128
//                                    double-buffered in LDS, consumed via ds_load)
//         LIF2 -> spk2_rec, mem2_rec
// ---------------------------------------------------------------------------

typedef __attribute__((ext_vector_type(16))) __bf16 v16bf;
typedef __attribute__((ext_vector_type(8)))  float  v8f;

#define B_SZ   256
#define NI     2048
#define NH     4096
#define NO     1024
#define TSTEPS 50
#define BETA   0.95f

union Frag16 {
    v16bf v;
    struct { uint4 a, b; } u;
};

static __device__ __forceinline__ unsigned short f32_to_bf16(float f) {
    unsigned int u = __float_as_uint(f);
    unsigned int r = u + 0x7FFFu + ((u >> 16) & 1u);   // round-to-nearest-even
    return (unsigned short)(r >> 16);
}

static __device__ __forceinline__ v8f wmma_bf16(v16bf a, v16bf b, v8f c) {
    return __builtin_amdgcn_wmma_f32_16x16x32_bf16(
        /*neg_a=*/false, a, /*neg_b=*/false, b,
        /*c_mod=*/(short)0, c, /*reuse_a=*/false, /*reuse_b=*/false);
}

// A fragment: 16x32 bf16, row-major source, tile origin (row0, k0).
// Lane L holds row M = L&15; half = L>>4 selects K-groups {0..15} / {16..31}.
static __device__ __forceinline__ v16bf load_a_frag(const unsigned short* __restrict__ base,
                                                    int ld, int row0, int k0, int lane) {
    int m = lane & 15, half = lane >> 4;
    const unsigned short* p = base + (size_t)(row0 + m) * (size_t)ld + (size_t)(k0 + half * 16);
    Frag16 f;
    f.u.a = *(const uint4*)(p);        // K = k0 + half*16 + (0..7)
    f.u.b = *(const uint4*)(p + 8);    // K = k0 + half*16 + (8..15)
    return f.v;
}

// B fragment: 32x16 bf16 where B[k][n] = W[n][k], W row-major [N, K] stride ld.
// Lanes 0-15 carry K = k0..k0+15, lanes 16-31 carry K = k0+16..k0+31; column N = lane&15.
static __device__ __forceinline__ v16bf load_b_frag(const unsigned short* __restrict__ base,
                                                    int ld, int col0, int k0, int lane) {
    int n = lane & 15, half = lane >> 4;
    const unsigned short* p = base + (size_t)(col0 + n) * (size_t)ld + (size_t)(k0 + half * 16);
    Frag16 f;
    f.u.a = *(const uint4*)(p);        // K = k0 + half*16 + (0..7)
    f.u.b = *(const uint4*)(p + 8);    // K = k0 + half*16 + (8..15)
    return f.v;
}

// B fragment from an LDS-resident row (64 bf16 K-contiguous elements per row).
static __device__ __forceinline__ v16bf load_b_lds(const unsigned short* p) {
    Frag16 f;
    f.u.a = *(const uint4*)(p);        // 16B ds_load
    f.u.b = *(const uint4*)(p + 8);    // 16B ds_load
    return f.v;
}

// ---------------------------------------------------------------------------
// Split f32 -> (bf16_hi, bf16_lo) so hi+lo ~= f32 (16-bit combined mantissa).
// ---------------------------------------------------------------------------
__global__ void snn_split_kernel(const float* __restrict__ src,
                                 unsigned short* __restrict__ hi,
                                 unsigned short* __restrict__ lo, int n) {
    int i = blockIdx.x * blockDim.x + threadIdx.x;
    if (i < n) {
        float f = src[i];
        unsigned short h = f32_to_bf16(f);
        float fh = __uint_as_float(((unsigned int)h) << 16);
        unsigned short l = f32_to_bf16(f - fh);
        hi[i] = h;
        lo[i] = l;
    }
}

// ---------------------------------------------------------------------------
// cur1 = x @ w1^T + b1 via 3-product bf16 split (runs once; direct-global).
// Wave tile: 16(M) x 64(N). 1024 waves -> 128 blocks x 256 threads.
// ---------------------------------------------------------------------------
__global__ __launch_bounds__(256) void snn_gemm_cur1_kernel(
    const unsigned short* __restrict__ xh, const unsigned short* __restrict__ xl,
    const unsigned short* __restrict__ w1h, const unsigned short* __restrict__ w1l,
    const float* __restrict__ b1, float* __restrict__ cur1) {
    int lane = threadIdx.x & 31;
    int wid  = threadIdx.x >> 5;
    int gw   = blockIdx.x * 8 + wid;
    int m0   = (gw & 15) << 4;
    int n0   = (gw >> 4) << 6;

    v8f acc[4] = {};
    for (int k0 = 0; k0 < NI; k0 += 32) {
        v16bf ah = load_a_frag(xh, NI, m0, k0, lane);
        v16bf al = load_a_frag(xl, NI, m0, k0, lane);
#pragma unroll
        for (int j = 0; j < 4; ++j) {
            v16bf bh = load_b_frag(w1h, NI, n0 + 16 * j, k0, lane);
            v16bf bl = load_b_frag(w1l, NI, n0 + 16 * j, k0, lane);
            acc[j] = wmma_bf16(ah, bh, acc[j]);
            acc[j] = wmma_bf16(al, bh, acc[j]);
            acc[j] = wmma_bf16(ah, bl, acc[j]);
        }
    }

    int half = lane >> 4, nn = lane & 15;
#pragma unroll
    for (int j = 0; j < 4; ++j) {
        int n = n0 + 16 * j + nn;
        float bias = b1[n];
#pragma unroll
        for (int r = 0; r < 8; ++r) {
            int m = m0 + r + 8 * half;
            cur1[(size_t)m * NH + n] = acc[j][r] + bias;
        }
    }
}

// ---------------------------------------------------------------------------
// LIF layer 1.
// ---------------------------------------------------------------------------
__global__ void snn_lif1_kernel(const float* __restrict__ cur1,
                                float* __restrict__ mem1,
                                unsigned short* __restrict__ spk1,
                                int first, int n) {
    int i = blockIdx.x * blockDim.x + threadIdx.x;
    if (i < n) {
        float mp = first ? 0.0f : mem1[i];
        float reset = (mp > 1.0f) ? 1.0f : 0.0f;
        float m = BETA * mp + cur1[i] - reset;
        mem1[i] = m;
        spk1[i] = (m > 1.0f) ? (unsigned short)0x3F80 : (unsigned short)0;  // bf16 1.0 / 0.0
    }
}

// ---------------------------------------------------------------------------
// cur2 = spk1 @ w2^T + b2 fused with LIF2.
// Block: 128 threads (4 waves), tile 64M x 64N -> 64 blocks.
// B (w2 hi+lo) staged into LDS with async global->LDS copies, double-buffered:
//   stage = 64N x 64K x {hi,lo} = 16 KB; 128 rows of 128 B; one row per thread
//   = 8 x global_load_async_to_lds_b128 per thread per stage (ASYNCcnt += 8/wave).
// ---------------------------------------------------------------------------
#define KSTAGE   64                 // K elements per stage
#define NSTAGES  (NH / KSTAGE)      // 64
#define ROWBYTES 128                // 64 bf16 per row

__global__ __launch_bounds__(128) void snn_gemm_lif2_kernel(
    const unsigned short* __restrict__ spk1,
    const unsigned short* __restrict__ w2h, const unsigned short* __restrict__ w2l,
    const float* __restrict__ b2, float* __restrict__ mem2,
    float* __restrict__ out_spk, float* __restrict__ out_mem, int first) {
    __shared__ unsigned short sh[2][128][KSTAGE];   // [buf][matrix*64 + n_local][k]

    const int tid  = threadIdx.x;
    const int lane = tid & 31;
    const int wid  = tid >> 5;               // 0..3
    const int mblk = blockIdx.x & 3;         // 4 M-blocks of 64 rows
    const int nblk = blockIdx.x >> 2;        // 16 N-blocks of 64 cols
    const int m0   = mblk * 64 + wid * 16;
    const int n0   = nblk * 64;

    // Per-thread staging source: row tid of the (128-row) hi|lo slab.
    const int mat  = tid >> 6;               // 0 = hi, 1 = lo
    const int nloc = tid & 63;
    const unsigned short* gsrc_base =
        (mat ? w2l : w2h) + (size_t)(n0 + nloc) * NH;   // + k0 per stage

    auto stage = [&](int st, int buf) {
        unsigned int lds = (unsigned int)(size_t)&sh[buf][tid][0];
        unsigned long long ga = (unsigned long long)(size_t)(gsrc_base + st * KSTAGE);
        // 8 x 16B async copies; IOFFSET applies to both LDS and global addresses.
        asm volatile("global_load_async_to_lds_b128 %0, %1, off"            :: "v"(lds), "v"(ga) : "memory");
        asm volatile("global_load_async_to_lds_b128 %0, %1, off offset:16"  :: "v"(lds), "v"(ga) : "memory");
        asm volatile("global_load_async_to_lds_b128 %0, %1, off offset:32"  :: "v"(lds), "v"(ga) : "memory");
        asm volatile("global_load_async_to_lds_b128 %0, %1, off offset:48"  :: "v"(lds), "v"(ga) : "memory");
        asm volatile("global_load_async_to_lds_b128 %0, %1, off offset:64"  :: "v"(lds), "v"(ga) : "memory");
        asm volatile("global_load_async_to_lds_b128 %0, %1, off offset:80"  :: "v"(lds), "v"(ga) : "memory");
        asm volatile("global_load_async_to_lds_b128 %0, %1, off offset:96"  :: "v"(lds), "v"(ga) : "memory");
        asm volatile("global_load_async_to_lds_b128 %0, %1, off offset:112" :: "v"(lds), "v"(ga) : "memory");
    };

    v8f acc[4] = {};
    const int nn = lane & 15, half = lane >> 4;

    stage(0, 0);
    for (int st = 0; st < NSTAGES; ++st) {
        if (st + 1 < NSTAGES) {
            stage(st + 1, (st + 1) & 1);
            asm volatile("s_wait_asynccnt 0x8" ::: "memory");   // stage st landed (in-order)
        } else {
            asm volatile("s_wait_asynccnt 0x0" ::: "memory");
        }
        __syncthreads();

        const int buf = st & 1;
#pragma unroll
        for (int s = 0; s < 2; ++s) {                 // two 32-K WMMA steps per stage
            v16bf a = load_a_frag(spk1, NH, m0, st * KSTAGE + s * 32, lane);
#pragma unroll
            for (int j = 0; j < 4; ++j) {
                const unsigned short* ph = &sh[buf][16 * j + nn][s * 32 + half * 16];
                const unsigned short* pl = &sh[buf][64 + 16 * j + nn][s * 32 + half * 16];
                v16bf bh = load_b_lds(ph);
                v16bf bl = load_b_lds(pl);
                acc[j] = wmma_bf16(a, bh, acc[j]);
                acc[j] = wmma_bf16(a, bl, acc[j]);
            }
        }
        __syncthreads();   // all reads of buf done before it is re-staged
    }

    // Fused LIF2 epilogue + recording.
#pragma unroll
    for (int j = 0; j < 4; ++j) {
        int n = n0 + 16 * j + nn;
        float bias = b2[n];
#pragma unroll
        for (int r = 0; r < 8; ++r) {
            int m = m0 + r + 8 * half;
            size_t idx = (size_t)m * NO + n;
            float cur = acc[j][r] + bias;
            float mp = first ? 0.0f : mem2[idx];
            float reset = (mp > 1.0f) ? 1.0f : 0.0f;
            float mnew = BETA * mp + cur - reset;
            mem2[idx] = mnew;
            out_spk[idx] = (mnew > 1.0f) ? 1.0f : 0.0f;
            out_mem[idx] = mnew;
        }
    }
}

// ---------------------------------------------------------------------------
extern "C" void kernel_launch(void* const* d_in, const int* in_sizes, int n_in,
                              void* d_out, int out_size, void* d_ws, size_t ws_size,
                              hipStream_t stream) {
    const float* x  = (const float*)d_in[0];   // [256, 2048]
    const float* w1 = (const float*)d_in[1];   // [4096, 2048]
    const float* b1 = (const float*)d_in[2];   // [4096]
    const float* w2 = (const float*)d_in[3];   // [1024, 4096]
    const float* b2 = (const float*)d_in[4];   // [1024]
    float* out = (float*)d_out;                // spk2_rec[50,256,1024] ++ mem2_rec[50,256,1024]

    char* ws = (char*)d_ws;
    size_t off = 0;
    auto alloc = [&](size_t bytes) -> void* {
        void* p = ws + off;
        off += (bytes + 255) & ~(size_t)255;
        return p;
    };

    float*          cur1 = (float*)alloc((size_t)B_SZ * NH * 4);
    float*          mem1 = (float*)alloc((size_t)B_SZ * NH * 4);
    float*          mem2 = (float*)alloc((size_t)B_SZ * NO * 4);
    unsigned short* spk1 = (unsigned short*)alloc((size_t)B_SZ * NH * 2);
    unsigned short* xh   = (unsigned short*)alloc((size_t)B_SZ * NI * 2);
    unsigned short* xl   = (unsigned short*)alloc((size_t)B_SZ * NI * 2);
    unsigned short* w1h  = (unsigned short*)alloc((size_t)NH * NI * 2);
    unsigned short* w1l  = (unsigned short*)alloc((size_t)NH * NI * 2);
    unsigned short* w2h  = (unsigned short*)alloc((size_t)NO * NH * 2);
    unsigned short* w2l  = (unsigned short*)alloc((size_t)NO * NH * 2);

    {
        int n = B_SZ * NI;
        snn_split_kernel<<<(n + 255) / 256, 256, 0, stream>>>(x, xh, xl, n);
        n = NH * NI;
        snn_split_kernel<<<(n + 255) / 256, 256, 0, stream>>>(w1, w1h, w1l, n);
        n = NO * NH;
        snn_split_kernel<<<(n + 255) / 256, 256, 0, stream>>>(w2, w2h, w2l, n);
    }

    snn_gemm_cur1_kernel<<<128, 256, 0, stream>>>(xh, xl, w1h, w1l, b1, cur1);

    const size_t step_elems = (size_t)B_SZ * NO;
    float* out_spk_base = out;
    float* out_mem_base = out + (size_t)TSTEPS * step_elems;

    for (int t = 0; t < TSTEPS; ++t) {
        int n1 = B_SZ * NH;
        snn_lif1_kernel<<<(n1 + 255) / 256, 256, 0, stream>>>(cur1, mem1, spk1, t == 0, n1);
        snn_gemm_lif2_kernel<<<64, 128, 0, stream>>>(
            spk1, w2h, w2l, b2, mem2,
            out_spk_base + (size_t)t * step_elems,
            out_mem_base + (size_t)t * step_elems, t == 0);
    }
}
